// TwoHeadGATSmall_71279277245016
// MI455X (gfx1250) — matrix operand order repaired and checked by reading, not compiled, not run on hardware
//
#include <hip/hip_runtime.h>
#include <hip/hip_bf16.h>

typedef _Float16 f16;
typedef __attribute__((ext_vector_type(16))) _Float16 v16h;
typedef __attribute__((ext_vector_type(8)))  float    v8f;

#define N_NODES 25000
#define N_EDGES 300000
#define DIN 32
#define EDF 16
#define HID 256
#define HEADS 4
#define ACT 16
#define GB 64
#define KW 1024

// ---------------- utility kernels ----------------

__global__ void fill_f32_kernel(float* p, float v, long n) {
  long i = (long)blockIdx.x * blockDim.x + threadIdx.x;
  long stride = (long)gridDim.x * blockDim.x;
  for (; i < n; i += stride) p[i] = v;
}

__global__ void f32_to_f16_kernel(const float* s, f16* d, long n) {
  long i = (long)blockIdx.x * blockDim.x + threadIdx.x;
  long stride = (long)gridDim.x * blockDim.x;
  for (; i < n; i += stride) d[i] = (f16)s[i];
}

// W [K][Nc] f32 row-major  ->  Wt [Nc][K] f16 (transposed, K contiguous)
__global__ void transpose_to_f16_kernel(const float* W, f16* Wt, int K, int Nc) {
  long idx = (long)blockIdx.x * blockDim.x + threadIdx.x;
  if (idx >= (long)K * Nc) return;
  int k = (int)(idx / Nc), n = (int)(idx % Nc);
  Wt[(long)n * K + k] = (f16)W[idx];
}

__device__ inline void atomicMaxF(float* addr, float v) {
  if (v >= 0.f) atomicMax((int*)addr, __float_as_int(v));
  else          atomicMin((unsigned int*)addr, __float_as_uint(v));
}

// ---------------- NNConv: edge encoder ----------------
// h_e = relu(edge_attr @ W1 + b1)  -> f16
// also accumulates in-degree and Xs[d] = sum_{e: dst=d} x[src[e]]  (for enc_b2 term)
__global__ void edge_enc_kernel(const float* __restrict__ ea, const float* __restrict__ w1,
                                const float* __restrict__ b1, const float* __restrict__ x,
                                const int* __restrict__ src, const int* __restrict__ dst,
                                f16* __restrict__ he, float* __restrict__ Xs,
                                float* __restrict__ deg, int E) {
  int e = blockIdx.x * blockDim.x + threadIdx.x;
  if (e >= E) return;
  float a[EDF];
#pragma unroll
  for (int i = 0; i < EDF; ++i) a[i] = ea[(long)e * EDF + i];
#pragma unroll 4
  for (int k = 0; k < 32; ++k) {
    float s = b1[k];
#pragma unroll
    for (int i = 0; i < EDF; ++i) s += a[i] * w1[i * 32 + k];
    he[(long)e * 32 + k] = (f16)fmaxf(s, 0.f);
  }
  int sN = src[e], dN = dst[e];
  atomicAdd(&deg[dN], 1.0f);
#pragma unroll 4
  for (int i = 0; i < 32; ++i) atomicAdd(&Xs[(long)dN * 32 + i], x[(long)sN * 32 + i]);
}

// ---------------- NNConv: msg GEMM (WMMA) + scatter ----------------
// msg = P @ B,  P[e, k*32+i] = he[e,k]*xh[src[e],i],  B = enc_w2 viewed [1024,32]
// b2t is B transposed: b2t[o][kk], kk contiguous. Each wave: one 16-edge M-tile,
// both N tiles (N=32). K chunks of 128 staged in LDS.
__global__ __launch_bounds__(256) void nnconv_msg_kernel(
    const f16* __restrict__ he, const f16* __restrict__ xh, const f16* __restrict__ b2t,
    const int* __restrict__ src, const int* __restrict__ dst,
    float* __restrict__ agg, int E) {
  __shared__ f16 P[8][16 * 128];
  const int lane = threadIdx.x & 31;
  const int w    = threadIdx.x >> 5;
  const long tile = (long)blockIdx.x * 8 + w;
  const int m  = lane & 15;
  const int hf = lane >> 4;
  v8f c0 = {};
  v8f c1 = {};
  for (int kc = 0; kc < 8; ++kc) {
    // build P chunk: 16 edges x 128 K values
    for (int t = lane; t < 16 * 128; t += 32) {
      int mm = t >> 7;
      int kl = t & 127;
      int kk = kc * 128 + kl;
      int k = kk >> 5, i = kk & 31;
      long e = tile * 16 + mm;
      f16 v = (f16)0.f;
      if (e < E) {
        int s = src[e];
        v = he[e * 32 + k] * xh[(long)s * 32 + i];
      }
      P[w][mm * 128 + kl] = v;
    }
    __syncthreads();
    for (int ks = 0; ks < 4; ++ks) {
      union { v16h v; unsigned u[8]; } A, B0, B1;
#pragma unroll
      for (int j = 0; j < 8; ++j) {
        // A fragment, this code's contiguous k order (same as B): VGPR j holds K = hf*16 + j*2 (+0,+1)
        int kl = ks * 32 + (hf << 4) + (j << 1);
        A.u[j] = *(const unsigned*)&P[w][m * 128 + kl];
        // B fragment, same contiguous k order as A: VGPR j holds K = hf*16 + j*2 (+0,+1), col = lane&15
        int kg = kc * 128 + ks * 32 + (hf << 4) + (j << 1);
        B0.u[j] = *(const unsigned*)&b2t[(long)m * 1024 + kg];
        B1.u[j] = *(const unsigned*)&b2t[(long)(m + 16) * 1024 + kg];
      }
      c0 = __builtin_amdgcn_wmma_f32_16x16x32_f16(false, A.v, false, B0.v, (short)0, c0, false, false);
      c1 = __builtin_amdgcn_wmma_f32_16x16x32_f16(false, A.v, false, B1.v, (short)0, c1, false, false);
    }
    __syncthreads();
  }
  // scatter C tile: row = hf*8+j, col = m; directly into agg via atomics
#pragma unroll
  for (int j = 0; j < 8; ++j) {
    long e = tile * 16 + hf * 8 + j;
    if (e < E) {
      int d = dst[e];
      atomicAdd(&agg[(long)d * 32 + m],      c0[j]);
      atomicAdd(&agg[(long)d * 32 + 16 + m], c1[j]);
    }
  }
}

// ---------------- NNConv epilogue: h0 = relu(mean_agg + x@root_w + bias) ----------------
__global__ void h0_kernel(const float* __restrict__ agg, const float* __restrict__ deg,
                          const float* __restrict__ Xs, const float* __restrict__ b2,
                          const float* __restrict__ x, const float* __restrict__ rootw,
                          const float* __restrict__ nbias, f16* __restrict__ h0h, int Nn) {
  int idx = blockIdx.x * blockDim.x + threadIdx.x;
  if (idx >= Nn * 32) return;
  int n = idx >> 5, o = idx & 31;
  float s = agg[idx];
#pragma unroll 4
  for (int i = 0; i < 32; ++i) s += Xs[(long)n * 32 + i] * b2[i * 32 + o];  // enc_b2 term
  s /= fmaxf(deg[n], 1.0f);
  float r = 0.f;
#pragma unroll 4
  for (int i = 0; i < 32; ++i) r += x[(long)n * 32 + i] * rootw[i * 32 + o];
  h0h[idx] = (f16)fmaxf(s + r + nbias[o], 0.f);
}

// ---------------- generic f16 WMMA GEMM: C[M,Nc] = A[M,K] @ Bt[Nc,K]^T ----------------
// Each wave computes a 16(M) x 64(N) strip: one shared A fragment per K-step,
// 4 B fragments, 4 WMMAs -> 4x A reuse, higher wmma/vmem density.
__global__ __launch_bounds__(256) void gemm_f16_kernel(
    const f16* __restrict__ A, const f16* __restrict__ Bt, f16* __restrict__ Ch,
    int M, int Nc, int K, long totalWaves) {
  const long wid = (long)blockIdx.x * 8 + (threadIdx.x >> 5);
  if (wid >= totalWaves) return;
  const int lane = threadIdx.x & 31;
  const int nsuper = Nc >> 6;                 // 64-column groups
  const long mtile = wid / nsuper;
  const int  ng = (int)(wid % nsuper);
  const int m  = lane & 15;
  const int hf = lane >> 4;
  int arow = (int)(mtile * 16 + m); if (arow >= M) arow = 0;  // clamp (rows guarded on store)
  v8f c[4] = {{}, {}, {}, {}};
  for (int kb = 0; kb < K; kb += 32) {
    union { v16h v; unsigned u[8]; } Af;
#pragma unroll
    for (int j = 0; j < 8; ++j) {
      int ka = kb + (hf << 4) + (j << 1);
      Af.u[j] = *(const unsigned*)&A[(long)arow * K + ka];
    }
#pragma unroll
    for (int t = 0; t < 4; ++t) {
      union { v16h v; unsigned u[8]; } Bf;
      const f16* brow = Bt + (long)(ng * 64 + t * 16 + m) * K + kb + (hf << 4);
#pragma unroll
      for (int j = 0; j < 8; ++j) Bf.u[j] = *(const unsigned*)&brow[j << 1];
      c[t] = __builtin_amdgcn_wmma_f32_16x16x32_f16(false, Af.v, false, Bf.v, (short)0, c[t], false, false);
    }
  }
#pragma unroll
  for (int t = 0; t < 4; ++t) {
#pragma unroll
    for (int j = 0; j < 8; ++j) {
      long r = mtile * 16 + hf * 8 + j;
      if (r < M) Ch[r * Nc + ng * 64 + t * 16 + m] = (f16)c[t][j];
    }
  }
}

// ---------------- GAT attention ----------------
__global__ void attn_coef_kernel(const f16* __restrict__ hh, const float* __restrict__ asrc,
                                 const float* __restrict__ adst, float* __restrict__ es,
                                 float* __restrict__ ed, int Nn) {
  int idx = blockIdx.x * blockDim.x + threadIdx.x;
  if (idx >= Nn * HEADS) return;
  int n = idx >> 2, h = idx & 3;
  const f16* row = hh + (long)n * 1024 + h * HID;
  float s1 = 0.f, s2 = 0.f;
  for (int d = 0; d < HID; ++d) {
    float v = (float)row[d];
    s1 += v * asrc[h * HID + d];
    s2 += v * adst[h * HID + d];
  }
  es[idx] = s1; ed[idx] = s2;
}

__global__ void attn_logit_kernel(const int* __restrict__ src, const int* __restrict__ dst,
                                  const float* __restrict__ es, const float* __restrict__ ed,
                                  float* __restrict__ ew, float* __restrict__ mx, int E, int E2) {
  int e = blockIdx.x * blockDim.x + threadIdx.x;
  if (e >= E2) return;
  int s, d;
  if (e < E) { s = src[e]; d = dst[e]; } else { s = d = e - E; }
#pragma unroll
  for (int h = 0; h < HEADS; ++h) {
    float v = es[s * HEADS + h] + ed[d * HEADS + h];
    v = v > 0.f ? v : 0.2f * v;                 // leaky_relu(0.2)
    ew[(long)e * HEADS + h] = v;
    atomicMaxF(&mx[d * HEADS + h], v);
  }
}

__global__ void attn_exp_kernel(const int* __restrict__ dst, const float* __restrict__ mx,
                                float* __restrict__ ew, float* __restrict__ den, int E, int E2) {
  int e = blockIdx.x * blockDim.x + threadIdx.x;
  if (e >= E2) return;
  int d = (e < E) ? dst[e] : e - E;
#pragma unroll
  for (int h = 0; h < HEADS; ++h) {
    float w = __expf(ew[(long)e * HEADS + h] - mx[d * HEADS + h]);
    ew[(long)e * HEADS + h] = w;
    atomicAdd(&den[d * HEADS + h], w);
  }
}

__global__ void attn_scatter_kernel(const int* __restrict__ src, const int* __restrict__ dst,
                                    const f16* __restrict__ hh, const float* __restrict__ ew,
                                    const float* __restrict__ den, float* __restrict__ osum,
                                    int E, int E2) {
  long idx = (long)blockIdx.x * blockDim.x + threadIdx.x;
  if (idx >= (long)E2 * HID) return;
  int e = (int)(idx >> 8), d2 = (int)(idx & 255);
  int s, d;
  if (e < E) { s = src[e]; d = dst[e]; } else { s = d = e - E; }
#pragma unroll
  for (int h = 0; h < HEADS; ++h) {
    float alpha = ew[(long)e * HEADS + h] / (den[d * HEADS + h] + 1e-16f);
    float v = (float)hh[(long)s * 1024 + h * HID + d2] * alpha;
    atomicAdd(&osum[(long)d * 1024 + h * HID + d2], v);
  }
}

__global__ void head_mean_kernel(const float* __restrict__ osum, const float* __restrict__ bias,
                                 f16* __restrict__ outH, float* __restrict__ outF, int Nn) {
  int idx = blockIdx.x * blockDim.x + threadIdx.x;
  if (idx >= Nn * HID) return;
  int n = idx >> 8, d2 = idx & 255;
  const float* r = osum + (long)n * 1024 + d2;
  float v = 0.25f * (r[0] + r[256] + r[512] + r[768]) + bias[d2];
  v = fmaxf(v, 0.f);
  outH[idx] = (f16)v;
  if (outF) outF[idx] = v;
}

// ---------------- heads ----------------
__global__ void wire_kernel(const float* __restrict__ h2, const int* __restrict__ wm,
                            const float* __restrict__ ww, const float* __restrict__ wb,
                            float* __restrict__ out) {
  int q = blockIdx.x * blockDim.x + threadIdx.x;
  if (q >= KW) return;
  int n = wm[q];
  float s = wb[0];
  for (int d = 0; d < HID; ++d) s += h2[(long)n * HID + d] * ww[d];
  out[q] = s;
}

__global__ void gcnt_kernel(const int* __restrict__ batch, float* __restrict__ gcnt, int Nn) {
  int n = blockIdx.x * blockDim.x + threadIdx.x;
  if (n >= Nn) return;
  atomicAdd(&gcnt[batch[n]], 1.0f);
}

__global__ void pool_kernel(const float* __restrict__ h2, const int* __restrict__ batch,
                            float* __restrict__ pooled, int Nn) {
  int idx = blockIdx.x * blockDim.x + threadIdx.x;
  if (idx >= Nn * HID) return;
  int n = idx >> 8, d = idx & 255;
  atomicAdd(&pooled[(long)batch[n] * HID + d], h2[idx]);
}

__global__ void action_kernel(const float* __restrict__ pooled, const float* __restrict__ gcnt,
                              const float* __restrict__ aw, const float* __restrict__ ab,
                              float* __restrict__ out) {
  int idx = blockIdx.x * blockDim.x + threadIdx.x;
  if (idx >= GB * ACT) return;
  int g = idx >> 4, a = idx & 15;
  float c = fmaxf(gcnt[g], 1.0f);
  float s = ab[a];
  for (int d = 0; d < HID; ++d) s += (pooled[(long)g * HID + d] / c) * aw[d * ACT + a];
  out[KW + idx] = s;
}

// ---------------- host ----------------
extern "C" void kernel_launch(void* const* d_in, const int* in_sizes, int n_in,
                              void* d_out, int out_size, void* d_ws, size_t ws_size,
                              hipStream_t stream) {
  const int N = N_NODES, E = N_EDGES, E2 = N_EDGES + N_NODES;
  const float* x        = (const float*)d_in[0];
  const float* ea       = (const float*)d_in[1];
  const int*   wm       = (const int*)  d_in[2];
  const int*   eidx     = (const int*)  d_in[3];
  const int*   batch    = (const int*)  d_in[4];
  const float* enc_w1   = (const float*)d_in[5];
  const float* enc_b1   = (const float*)d_in[6];
  const float* enc_w2   = (const float*)d_in[7];
  const float* enc_b2   = (const float*)d_in[8];
  const float* root_w   = (const float*)d_in[9];
  const float* nn_bias  = (const float*)d_in[10];
  const float* g1_w     = (const float*)d_in[11];
  const float* g1_as    = (const float*)d_in[12];
  const float* g1_ad    = (const float*)d_in[13];
  const float* g1_b     = (const float*)d_in[14];
  const float* g2_w     = (const float*)d_in[15];
  const float* g2_as    = (const float*)d_in[16];
  const float* g2_ad    = (const float*)d_in[17];
  const float* g2_b     = (const float*)d_in[18];
  const float* wire_w   = (const float*)d_in[19];
  const float* wire_b   = (const float*)d_in[20];
  const float* act_w    = (const float*)d_in[21];
  const float* act_b    = (const float*)d_in[22];
  const int* src = eidx;
  const int* dst = eidx + E;
  float* out = (float*)d_out;

  char* ws = (char*)d_ws;
  size_t off = 0;
  auto alloc = [&](size_t bytes) -> char* {
    off = (off + 255) & ~(size_t)255;
    char* p = ws + off;
    off += bytes;
    return p;
  };
  f16*   xh     = (f16*)  alloc((size_t)N * 32 * 2);
  f16*   he     = (f16*)  alloc((size_t)E * 32 * 2);
  f16*   b2t    = (f16*)  alloc((size_t)32 * 1024 * 2);
  f16*   w1t    = (f16*)  alloc((size_t)1024 * 32 * 2);
  f16*   w2t    = (f16*)  alloc((size_t)1024 * 256 * 2);
  float* Xs     = (float*)alloc((size_t)N * 32 * 4);
  float* agg    = (float*)alloc((size_t)N * 32 * 4);
  float* deg    = (float*)alloc((size_t)N * 4);
  f16*   h0h    = (f16*)  alloc((size_t)N * 32 * 2);
  f16*   hh     = (f16*)  alloc((size_t)N * 1024 * 2);
  float* es     = (float*)alloc((size_t)N * 4 * 4);
  float* ed     = (float*)alloc((size_t)N * 4 * 4);
  float* mx     = (float*)alloc((size_t)N * 4 * 4);
  float* den    = (float*)alloc((size_t)N * 4 * 4);
  float* ewb    = (float*)alloc((size_t)E2 * 4 * 4);
  float* osum   = (float*)alloc((size_t)N * 1024 * 4);
  f16*   h1h    = (f16*)  alloc((size_t)N * 256 * 2);
  float* h2     = (float*)alloc((size_t)N * 256 * 4);
  float* pooled = (float*)alloc((size_t)GB * 256 * 4);
  float* gcnt   = (float*)alloc((size_t)GB * 4);

  auto cdiv = [](long a, long b) { return (int)((a + b - 1) / b); };

  // ---- prep: conversions / transposes / zeroing ----
  f32_to_f16_kernel<<<2048, 256, 0, stream>>>(x, xh, (long)N * 32);
  transpose_to_f16_kernel<<<cdiv(1024L * 32, 256), 256, 0, stream>>>(enc_w2, b2t, 1024, 32);
  transpose_to_f16_kernel<<<cdiv(32L * 1024, 256), 256, 0, stream>>>(g1_w, w1t, 32, 1024);
  transpose_to_f16_kernel<<<cdiv(256L * 1024, 256), 256, 0, stream>>>(g2_w, w2t, 256, 1024);
  fill_f32_kernel<<<2048, 256, 0, stream>>>(agg, 0.f, (long)N * 32);
  fill_f32_kernel<<<2048, 256, 0, stream>>>(Xs, 0.f, (long)N * 32);
  fill_f32_kernel<<<2048, 256, 0, stream>>>(deg, 0.f, (long)N);

  // ---- NNConv ----
  edge_enc_kernel<<<cdiv(E, 256), 256, 0, stream>>>(ea, enc_w1, enc_b1, x, src, dst, he, Xs, deg, E);
  {
    int mtiles = (E + 15) / 16;                  // 18750
    nnconv_msg_kernel<<<cdiv(mtiles, 8), 256, 0, stream>>>(he, xh, b2t, src, dst, agg, E);
  }
  h0_kernel<<<cdiv((long)N * 32, 256), 256, 0, stream>>>(agg, deg, Xs, enc_b2, x, root_w, nn_bias, h0h, N);

  // ---- GAT layers ----
  auto run_gat = [&](const f16* Ain, int K, const float* asrc, const float* adst,
                     const float* bias, f16* outH, float* outF) {
    long mtiles = (N + 15) / 16;
    long waves = mtiles * (1024 / 64);           // 64-wide N strips per wave
    gemm_f16_kernel<<<cdiv(waves, 8), 256, 0, stream>>>(Ain, (K == 32) ? w1t : w2t, hh, N, 1024, K, waves);
    attn_coef_kernel<<<cdiv((long)N * HEADS, 256), 256, 0, stream>>>(hh, asrc, adst, es, ed, N);
    fill_f32_kernel<<<256, 256, 0, stream>>>(mx, -1e30f, (long)N * HEADS);
    fill_f32_kernel<<<256, 256, 0, stream>>>(den, 0.f, (long)N * HEADS);
    attn_logit_kernel<<<cdiv(E2, 256), 256, 0, stream>>>(src, dst, es, ed, ewb, mx, E, E2);
    attn_exp_kernel<<<cdiv(E2, 256), 256, 0, stream>>>(dst, mx, ewb, den, E, E2);
    fill_f32_kernel<<<4096, 256, 0, stream>>>(osum, 0.f, (long)N * 1024);
    attn_scatter_kernel<<<cdiv((long)E2 * HID, 256), 256, 0, stream>>>(src, dst, hh, ewb, den, osum, E, E2);
    head_mean_kernel<<<cdiv((long)N * HID, 256), 256, 0, stream>>>(osum, bias, outH, outF, N);
  };
  run_gat(h0h, 32,  g1_as, g1_ad, g1_b, h1h, nullptr);   // h1 (f16)
  run_gat(h1h, 256, g2_as, g2_ad, g2_b, h1h, h2);        // h2 (f32 for heads)

  // ---- heads ----
  wire_kernel<<<cdiv(KW, 256), 256, 0, stream>>>(h2, wm, wire_w, wire_b, out);
  fill_f32_kernel<<<64, 256, 0, stream>>>(pooled, 0.f, (long)GB * HID);
  fill_f32_kernel<<<1, 64, 0, stream>>>(gcnt, 0.f, (long)GB);
  gcnt_kernel<<<cdiv(N, 256), 256, 0, stream>>>(batch, gcnt, N);
  pool_kernel<<<cdiv((long)N * HID, 256), 256, 0, stream>>>(h2, batch, pooled, N);
  action_kernel<<<cdiv(GB * ACT, 256), 256, 0, stream>>>(pooled, gcnt, act_w, act_b, out);
}
